// Flow_1322849927662
// MI455X (gfx1250) — compile-verified, matrix operand order checked by reading, not run on hardware
//
#include <hip/hip_runtime.h>
#include <hip/hip_bf16.h>

typedef __attribute__((ext_vector_type(16))) _Float16 v16h;
typedef __attribute__((ext_vector_type(8)))  float    v8f;

#define DIMS_IN 16
#define BINS    10
#define UNITS   32
#define BLOCKS  8
#define POUT    248          // 8 * 31
#define POUT_PAD 256
#define BOUND   10.0f
#define MIN_W   0.001f
#define MIN_H   0.001f
#define MIN_D   0.001f

#define NWAVES  4            // waves per workgroup (wave32)
#define SPW     16           // samples per wave (WMMA M)
#define WG_SAMPLES (NWAVES * SPW)   // 64
#define TPB     (NWAVES * 32)       // 128

// mask bit p of dim index; even blocks: cond = bit set, odd blocks: inverted
__constant__ int CONDI[BLOCKS][8] = {
    {1,3,5,7,9,11,13,15}, {0,2,4,6,8,10,12,14},
    {2,3,6,7,10,11,14,15}, {0,1,4,5,8,9,12,13},
    {4,5,6,7,12,13,14,15}, {0,1,2,3,8,9,10,11},
    {8,9,10,11,12,13,14,15}, {0,1,2,3,4,5,6,7}
};
__constant__ int TRAFOI[BLOCKS][8] = {
    {0,2,4,6,8,10,12,14}, {1,3,5,7,9,11,13,15},
    {0,1,4,5,8,9,12,13}, {2,3,6,7,10,11,14,15},
    {0,1,2,3,8,9,10,11}, {4,5,6,7,12,13,14,15},
    {0,1,2,3,4,5,6,7}, {8,9,10,11,12,13,14,15}
};

// CDNA5 16-bit A/B fragment K-map (ISA 7.12.2, 16-bit A 16x32):
// lane group g = lane>>4; element h (v16h index) -> VGPR v=h>>1, slot s=h&1
// K = (v<4 ? 0 : 16) + g*8 + (v&3)*2 + s
__device__ __forceinline__ int kmap(int h, int g) {
    int v = h >> 1, s = h & 1;
    return ((v < 4) ? 0 : 16) + g * 8 + ((v & 3) << 1) + s;
}

// ---- dynamic LDS layout (bytes) ----
#define OFF_W0 0                                   // 2 tiles * 32 lanes * 16 halves
#define OFF_W1 (OFF_W0 + 1024*2)
#define OFF_W2 (OFF_W1 + 1024*2)
#define OFF_WO (OFF_W2 + 1024*2)                   // 16 tiles * 32 * 16 halves
#define OFF_B0 (OFF_WO + 8192*2)
#define OFF_B1 (OFF_B0 + 32*4)
#define OFF_B2 (OFF_B1 + 32*4)
#define OFF_BO (OFF_B2 + 32*4)                     // 256 f32 (padded)
#define OFF_X  (OFF_BO + 256*4)                    // 64 samples * stride 17 f32
#define OFF_H  (OFF_X  + WG_SAMPLES*17*4)          // NWAVES * 16 * stride 33 f32
#define OFF_P  (OFF_H  + NWAVES*16*33*4)           // NWAVES * 16 * stride 264 f32
#define OFF_J  (OFF_P  + NWAVES*16*264*4)          // 64 f32
#define SMEM_BYTES (OFF_J + WG_SAMPLES*4)

extern "C" __global__ __launch_bounds__(TPB, 1)
void flow_fused_kernel(const float* __restrict__ xg,
                       const float* __restrict__ W0g, const float* __restrict__ b0g,
                       const float* __restrict__ W1g, const float* __restrict__ b1g,
                       const float* __restrict__ W2g, const float* __restrict__ b2g,
                       const float* __restrict__ Wog, const float* __restrict__ bog,
                       float* __restrict__ out, int nsamp)
{
    extern __shared__ __align__(32) char smem[];
    _Float16* w0 = (_Float16*)(smem + OFF_W0);
    _Float16* w1 = (_Float16*)(smem + OFF_W1);
    _Float16* w2 = (_Float16*)(smem + OFF_W2);
    _Float16* wo = (_Float16*)(smem + OFF_WO);
    float* bs0 = (float*)(smem + OFF_B0);
    float* bs1 = (float*)(smem + OFF_B1);
    float* bs2 = (float*)(smem + OFF_B2);
    float* bso = (float*)(smem + OFF_BO);
    float* xs  = (float*)(smem + OFF_X);    // [64][17]
    float* hs  = (float*)(smem + OFF_H);    // [NWAVES][16][33]
    float* ps  = (float*)(smem + OFF_P);    // [NWAVES][16][264]
    float* js  = (float*)(smem + OFF_J);    // [64]

    const int tid  = threadIdx.x;
    const int wave = tid >> 5;
    const int lane = tid & 31;
    const int g    = lane >> 4;     // A/B fragment lane group
    const int col  = lane & 15;     // D fragment column / A row

    const int base = blockIdx.x * WG_SAMPLES;

    // ---- load x tile (64 samples x 16 dims), init jac accumulators ----
    for (int i = tid; i < WG_SAMPLES * DIMS_IN; i += TPB) {
        int s = i >> 4, d = i & 15;
        xs[s * 17 + d] = xg[(size_t)(base + s) * DIMS_IN + d];
    }
    if (tid < WG_SAMPLES) js[tid] = 0.f;

    float* hrow = hs + wave * 16 * 33;
    float* prow = ps + wave * 16 * 264;
    float* xwav = xs + wave * 16 * 17;

    for (int blk = 0; blk < BLOCKS; ++blk) {
        __syncthreads();   // previous block done reading weights (also fences x-load, js init)

        // ---- stage this block's weights into WMMA-B fragment order (f16) ----
        for (int i = tid; i < 1024; i += TPB) {
            int t = i >> 9, l = (i >> 4) & 31, h = i & 15;
            int k = kmap(h, l >> 4);
            int n = t * 16 + (l & 15);
            w0[i] = (_Float16)((k < 8) ? W0g[blk * 8 * UNITS + k * UNITS + n] : 0.f);
            w1[i] = (_Float16)W1g[blk * UNITS * UNITS + k * UNITS + n];
            w2[i] = (_Float16)W2g[blk * UNITS * UNITS + k * UNITS + n];
        }
        for (int i = tid; i < 8192; i += TPB) {
            int t = i >> 9, l = (i >> 4) & 31, h = i & 15;
            int k = kmap(h, l >> 4);
            int n = t * 16 + (l & 15);
            wo[i] = (_Float16)((n < POUT) ? Wog[blk * UNITS * POUT + k * POUT + n] : 0.f);
        }
        if (tid < UNITS) {
            bs0[tid] = b0g[blk * UNITS + tid];
            bs1[tid] = b1g[blk * UNITS + tid];
            bs2[tid] = b2g[blk * UNITS + tid];
        }
        for (int i = tid; i < POUT_PAD; i += TPB)
            bso[i] = (i < POUT) ? bog[blk * POUT + i] : 0.f;

        __syncthreads();

        const v8f cz = {0.f,0.f,0.f,0.f,0.f,0.f,0.f,0.f};
        v16h a;

        // ---- layer 0: x_cond (K pad 8->32) @ W0 -> h ----
        {
            const float* xrow = xwav + col * 17;
            #pragma unroll
            for (int h = 0; h < 16; ++h) {
                int k = kmap(h, g);
                float xv = xrow[CONDI[blk][k & 7]];
                a[h] = (k < 8) ? (_Float16)xv : (_Float16)0.f;
            }
            const v16h* wv = (const v16h*)w0;
            #pragma unroll
            for (int t = 0; t < 2; ++t) {
                v16h b = wv[t * 32 + lane];
                v8f c = __builtin_amdgcn_wmma_f32_16x16x32_f16(false, a, false, b, (short)0, cz, false, false);
                int N = t * 16 + col;
                float bias = bs0[N];
                #pragma unroll
                for (int r = 0; r < 8; ++r)
                    hrow[(r + 8 * g) * 33 + N] = fmaxf(c[r] + bias, 0.f);
            }
        }

        // ---- layers 1, 2: 32x32 ----
        const _Float16* wlay[2] = { w1, w2 };
        const float*    blay[2] = { bs1, bs2 };
        #pragma unroll
        for (int L = 0; L < 2; ++L) {
            #pragma unroll
            for (int h = 0; h < 16; ++h)
                a[h] = (_Float16)hrow[col * 33 + kmap(h, g)];
            const v16h* wv = (const v16h*)wlay[L];
            #pragma unroll
            for (int t = 0; t < 2; ++t) {
                v16h b = wv[t * 32 + lane];
                v8f c = __builtin_amdgcn_wmma_f32_16x16x32_f16(false, a, false, b, (short)0, cz, false, false);
                int N = t * 16 + col;
                float bias = blay[L][N];
                #pragma unroll
                for (int r = 0; r < 8; ++r)
                    hrow[(r + 8 * g) * 33 + N] = fmaxf(c[r] + bias, 0.f);
            }
        }

        // ---- output layer: 32 -> 248 (N pad 256), no relu ----
        {
            #pragma unroll
            for (int h = 0; h < 16; ++h)
                a[h] = (_Float16)hrow[col * 33 + kmap(h, g)];
            const v16h* wv = (const v16h*)wo;
            #pragma unroll
            for (int t = 0; t < 16; ++t) {
                v16h b = wv[t * 32 + lane];
                v8f c = __builtin_amdgcn_wmma_f32_16x16x32_f16(false, a, false, b, (short)0, cz, false, false);
                int N = t * 16 + col;
                float bias = bso[N];
                #pragma unroll
                for (int r = 0; r < 8; ++r)
                    prow[(r + 8 * g) * 264 + N] = c[r] + bias;   // cols 248..255 are padding
            }
        }

        // ---- RQ spline: 128 (sample,dim) pairs per wave, 4 per lane ----
        #pragma unroll
        for (int q = 0; q < 4; ++q) {
            int pidx = lane + 32 * q;
            int m = pidx & 15;
            int j = pidx >> 4;
            const float* pp = prow + m * 264 + j * 31;
            float P[31];
            #pragma unroll
            for (int k = 0; k < 31; ++k) P[k] = pp[k];

            // widths softmax -> cumwidths
            float mx = P[0];
            #pragma unroll
            for (int k = 1; k < BINS; ++k) mx = fmaxf(mx, P[k]);
            float e[10]; float sum = 0.f;
            #pragma unroll
            for (int k = 0; k < BINS; ++k) { e[k] = __expf(P[k] - mx); sum += e[k]; }
            float cw[11]; cw[0] = -BOUND;
            float sc = (1.f - MIN_W * BINS) / sum, acc = 0.f;
            #pragma unroll
            for (int k = 0; k < BINS; ++k) { acc += MIN_W + e[k] * sc; cw[k + 1] = -BOUND + 2.f * BOUND * acc; }
            cw[10] = BOUND;

            // heights softmax -> cumheights
            mx = P[10];
            #pragma unroll
            for (int k = 1; k < BINS; ++k) mx = fmaxf(mx, P[10 + k]);
            sum = 0.f;
            #pragma unroll
            for (int k = 0; k < BINS; ++k) { e[k] = __expf(P[10 + k] - mx); sum += e[k]; }
            float chh[11]; chh[0] = -BOUND;
            sc = (1.f - MIN_H * BINS) / sum; acc = 0.f;
            #pragma unroll
            for (int k = 0; k < BINS; ++k) { acc += MIN_H + e[k] * sc; chh[k + 1] = -BOUND + 2.f * BOUND * acc; }
            chh[10] = BOUND;

            // derivatives: MIN_D + softplus
            float dd[11];
            #pragma unroll
            for (int k = 0; k < BINS + 1; ++k) {
                float z = P[20 + k];
                dd[k] = MIN_D + fmaxf(z, 0.f) + log1pf(__expf(-fabsf(z)));
            }

            int td = TRAFOI[blk][j];
            float xv = xwav[m * 17 + td];
            bool inside = (xv >= -BOUND) && (xv <= BOUND);
            float xc = fminf(fmaxf(xv, -BOUND), BOUND);

            int bi = 0;
            #pragma unroll
            for (int k = 1; k < BINS; ++k) bi += (xc >= cw[k]) ? 1 : 0;

            float x0 = 0.f, w = 1.f, y0 = 0.f, hh = 1.f, d0 = 1.f, d1 = 1.f;
            #pragma unroll
            for (int k = 0; k < BINS; ++k) {
                if (bi == k) {
                    x0 = cw[k]; w = cw[k + 1] - cw[k];
                    y0 = chh[k]; hh = chh[k + 1] - chh[k];
                    d0 = dd[k]; d1 = dd[k + 1];
                }
            }
            float delta = hh / w;
            float t   = (xc - x0) / w;
            float t1m = t * (1.f - t);
            float den = delta + (d0 + d1 - 2.f * delta) * t1m;
            float y   = y0 + hh * (delta * t * t + d0 * t1m) / den;
            float omt = 1.f - t;
            float num2 = d1 * t * t + 2.f * delta * t1m + d0 * omt * omt;
            float ld  = __logf(delta * delta * num2) - 2.f * __logf(den);

            xwav[m * 17 + td] = inside ? y : xv;
            if (inside) atomicAdd(&js[wave * 16 + m], ld);
        }
    }

    __syncthreads();
    if (tid < WG_SAMPLES) out[base + tid] = js[tid];
}

extern "C" void kernel_launch(void* const* d_in, const int* in_sizes, int n_in,
                              void* d_out, int out_size, void* d_ws, size_t ws_size,
                              hipStream_t stream) {
    (void)n_in; (void)out_size; (void)d_ws; (void)ws_size;
    const float* xg  = (const float*)d_in[0];
    const float* W0g = (const float*)d_in[1];
    const float* b0g = (const float*)d_in[2];
    const float* W1g = (const float*)d_in[3];
    const float* b1g = (const float*)d_in[4];
    const float* W2g = (const float*)d_in[5];
    const float* b2g = (const float*)d_in[6];
    const float* Wog = (const float*)d_in[7];
    const float* bog = (const float*)d_in[8];
    float* out = (float*)d_out;

    int nsamp = in_sizes[0] / DIMS_IN;          // 262144
    int nblocks = nsamp / WG_SAMPLES;           // 4096 (divisible)
    flow_fused_kernel<<<dim3(nblocks), dim3(TPB), SMEM_BYTES, stream>>>(
        xg, W0g, b0g, W1g, b1g, W2g, b2g, Wog, bog, out, nsamp);
}
